// Decoder_61667140436156
// MI455X (gfx1250) — matrix operand order repaired and checked by reading, not compiled, not run on hardware
//
#include <hip/hip_runtime.h>
#include <hip/hip_bf16.h>
#include <stdint.h>

// MI455X / gfx1250, wave32. GEMMs via v_wmma_f32_16x16x32_bf16 (f32 acc).
// Big tile loads staged to LDS with the Tensor Data Mover (TDM); this
// toolchain exposes the 6-arg clang-23 builtin form.

typedef __attribute__((ext_vector_type(16))) __bf16 v16bf;
typedef __attribute__((ext_vector_type(8)))  float  v8f;
typedef __attribute__((ext_vector_type(4)))  unsigned int u32x4;
typedef __attribute__((ext_vector_type(8)))  int    i32x8;
typedef __attribute__((ext_vector_type(4)))  int    i32x4;

#define BATCH 64
#define EMB   128
#define HID   256
#define VOCAB 50000
#define SLEN  400

#if __has_builtin(__builtin_amdgcn_tensor_load_to_lds) && \
    __has_builtin(__builtin_amdgcn_s_wait_tensorcnt)
#define USE_TDM 1
#else
#define USE_TDM 0
#endif

#if USE_TDM
// ---------------------------------------------------------------------------
// Issue one 2D TDM load: tile_h rows x tile_w f32 elements, row stride
// stride_w elements, from global tile start `gaddr` into LDS offset lds_off.
// D# layout per CDNA5 ISA (group0 128b, group1
// 256b; groups 2/3 zero for 2D tiles). Uniform args -> SGPR descriptor.
// 6-arg builtin form: (g0, g1, g2, g3, g_extra, cpol).
// ---------------------------------------------------------------------------
__device__ __forceinline__ void tdm_load_2d_f32(unsigned lds_off,
                                                const void* gaddr,
                                                unsigned tensor_w,
                                                unsigned tensor_h,
                                                unsigned stride_w,
                                                unsigned tile_w,
                                                unsigned tile_h) {
  const unsigned long long ga = (unsigned long long)(uintptr_t)gaddr;
  u32x4 g0;
  g0[0] = 1u;                                   // count=1 (valid), user mode
  g0[1] = lds_off;                              // lds_addr [63:32]
  g0[2] = (unsigned)(ga & 0xFFFFFFFFull);       // global_addr [95:64]
  g0[3] = (unsigned)((ga >> 32) & 0x01FFFFFFull) | (2u << 30); // [120:96]+type=2
  i32x8 g1;
  g1[0] = (int)(2u << 16);                      // data_size=2 (4 bytes)
  g1[1] = (int)((tensor_w & 0xFFFFu) << 16);    // tensor_dim0[15:0] at [63:48]
  g1[2] = (int)(((tensor_w >> 16) & 0xFFFFu) |  // tensor_dim0[31:16]
                ((tensor_h & 0xFFFFu) << 16));  // tensor_dim1[15:0]
  g1[3] = (int)(((tensor_h >> 16) & 0xFFFFu) |  // tensor_dim1[31:16]
                ((tile_w & 0xFFFFu) << 16));    // tile_dim0
  g1[4] = (int)(tile_h & 0xFFFFu);              // tile_dim1 (tile_dim2 = 0)
  g1[5] = (int)stride_w;                        // tensor_dim0_stride[31:0]
  g1[6] = 0;                                    // stride hi / dim1_stride lo
  g1[7] = 0;
  const i32x4 z4 = {0, 0, 0, 0};
  const i32x8 z8 = {0, 0, 0, 0, 0, 0, 0, 0};
  __builtin_amdgcn_tensor_load_to_lds(g0, g1, z4, z4, z8, 0);
}
#endif

// ---------------------------------------------------------------------------
// One 16x16 f32 tile accumulated over K (multiple of 32) with bf16 WMMA.
// Per-lane element mapping used here (contiguous k, not the ISA table order; 16-bit, wave32):
//   A (16xK): m = lane&15, half = lane>>4; elem i -> K = k0 + 16*half + i
//   B (Kx16): n = lane&15, half = lane>>4; elem i -> K = k0 + 16*half + i
//   C/D     : n = lane&15; vgpr r -> M = r + 8*half
// ---------------------------------------------------------------------------
template <typename LA, typename LB>
__device__ __forceinline__ v8f bf16_tile_mma(int K, v8f acc, LA loadA, LB loadB) {
  const int lane = threadIdx.x & 31;
  const int half = lane >> 4;
  for (int k0 = 0; k0 < K; k0 += 32) {
    v16bf a, b;
#pragma unroll
    for (int i = 0; i < 16; ++i) {
      a[i] = loadA(k0 + 16 * half + i);
      b[i] = loadB(k0 + 16 * half + i);
    }
    acc = __builtin_amdgcn_wmma_f32_16x16x32_bf16(false, a, false, b, (short)0,
                                                  acc, false, false);
  }
  return acc;
}

__device__ __forceinline__ float sigmoidf_(float x) {
  return 1.0f / (1.0f + __expf(-x));
}

// ---------------------------------------------------------------------------
// K1a: gates = embed[word] @ W_ih^T + hidden @ W_hh^T   (biases added later)
// 64x1024 output, tiles 4(M) x 64(N) = 256; 8 blocks x 8 waves, 4 tiles/wave.
// ---------------------------------------------------------------------------
__global__ void k_gates(const int* __restrict__ word,
                        const float* __restrict__ hidden,
                        const float* __restrict__ embed,
                        const float* __restrict__ W_ih,
                        const float* __restrict__ W_hh,
                        float* __restrict__ gates) {
  const int gw   = blockIdx.x * (blockDim.x >> 5) + (threadIdx.x >> 5);
  const int lane = threadIdx.x & 31;
  const int ml   = lane & 15;
  const int half = lane >> 4;
  for (int it = 0; it < 4; ++it) {
    const int t  = gw * 4 + it;        // 0..255
    const int mt = t & 3;              // 4 M-tiles
    const int nt = t >> 2;             // 64 N-tiles
    const int gm = mt * 16 + ml;
    const int gn = nt * 16 + ml;
    const float* xrow = embed + (size_t)word[gm] * EMB;
    const float* hrow = hidden + (size_t)gm * HID;
    const float* wih  = W_ih + (size_t)gn * EMB;   // gates = x @ W_ih^T
    const float* whh  = W_hh + (size_t)gn * HID;
    v8f acc = {};
    acc = bf16_tile_mma(EMB, acc,
                        [&](int k) { return (__bf16)xrow[k]; },
                        [&](int k) { return (__bf16)wih[k]; });
    acc = bf16_tile_mma(HID, acc,
                        [&](int k) { return (__bf16)hrow[k]; },
                        [&](int k) { return (__bf16)whh[k]; });
#pragma unroll
    for (int r = 0; r < 8; ++r)
      gates[(size_t)(mt * 16 + r + 8 * half) * (4 * HID) + gn] = acc[r];
  }
}

// ---------------------------------------------------------------------------
// K1b: LSTM pointwise. Writes c_t and tanh(h_t) to d_out and bf16 h_t into
// the cat staging buffer (cols 512..767).
// ---------------------------------------------------------------------------
__global__ void k_lstm_pw(const float* __restrict__ gates,
                          const float* __restrict__ cell,
                          const float* __restrict__ b_ih,
                          const float* __restrict__ b_hh,
                          float* __restrict__ out_ht,
                          float* __restrict__ out_ct,
                          __bf16* __restrict__ cat_bf) {
  const int idx = blockIdx.x * blockDim.x + threadIdx.x;  // 0..16383
  const int b = idx >> 8, j = idx & 255;
  const size_t g0 = (size_t)b * (4 * HID);
  const float gi = gates[g0 + j]           + b_ih[j]           + b_hh[j];
  const float gf = gates[g0 + HID + j]     + b_ih[HID + j]     + b_hh[HID + j];
  const float gg = gates[g0 + 2 * HID + j] + b_ih[2 * HID + j] + b_hh[2 * HID + j];
  const float go = gates[g0 + 3 * HID + j] + b_ih[3 * HID + j] + b_hh[3 * HID + j];
  const float i_ = sigmoidf_(gi);
  const float f_ = sigmoidf_(gf);
  const float g_ = tanhf(gg);
  const float o_ = sigmoidf_(go);
  const float ct = f_ * cell[idx] + i_ * g_;
  const float ht = tanhf(o_ * tanhf(ct));   // extra tanh per reference
  out_ct[idx] = ct;
  out_ht[idx] = ht;
  cat_bf[(size_t)b * (3 * HID) + 2 * HID + j] = (__bf16)ht;
}

// ---------------------------------------------------------------------------
// K1c: hproj = h_t @ W_h^T + attn_b  (64x256, K=256). W_h = attn_W[:,512:768].
// ---------------------------------------------------------------------------
__global__ void k_hproj(const __bf16* __restrict__ cat_bf,
                        const float* __restrict__ attn_W,
                        const float* __restrict__ attn_b,
                        float* __restrict__ hproj) {
  const int w    = threadIdx.x >> 5;
  const int lane = threadIdx.x & 31;
  const int ml   = lane & 15;
  const int half = lane >> 4;
  for (int it = 0; it < 4; ++it) {
    const int t  = w * 4 + it;
    const int mt = t & 3, nt = t >> 2;
    const int gm = mt * 16 + ml, gn = nt * 16 + ml;
    const __bf16* arow = cat_bf + (size_t)gm * (3 * HID) + 2 * HID;
    const float*  bcol = attn_W + (size_t)gn * (3 * HID) + 2 * HID;
    v8f acc = {};
    acc = bf16_tile_mma(HID, acc,
                        [&](int k) { return arow[k]; },
                        [&](int k) { return (__bf16)bcol[k]; });
    const float bias = attn_b[gn];
#pragma unroll
    for (int r = 0; r < 8; ++r)
      hproj[(size_t)(mt * 16 + r + 8 * half) * HID + gn] = acc[r] + bias;
  }
}

// ---------------------------------------------------------------------------
// K2: e[b,s] = tanh(enc[b,s,:] @ W_enc^T + hproj[b,:]) . v
// Block = (s-tile, b). The 16x512 enc tile is DMA'd into LDS by the TDM
// (wave 0 issues tensor_load_to_lds + s_wait_tensorcnt), then 4 waves cover
// the 16 N-tiles of the 16x256 pre tile; LDS atomic-add reduction over N.
// ---------------------------------------------------------------------------
__global__ void k_scores(const float* __restrict__ enc,
                         const float* __restrict__ attn_W,
                         const float* __restrict__ hproj,
                         const float* __restrict__ vvec,
                         float* __restrict__ e_out) {
  __shared__ float encF[16 * 2 * HID];   // 32 KB f32 tile
  __shared__ float e_acc[16];
  const int st  = blockIdx.x;            // 25 s-tiles
  const int b   = blockIdx.y;            // 64 batches
  const int tid = threadIdx.x;           // 128 threads / 4 waves
  const float* src = enc + ((size_t)b * SLEN + (size_t)st * 16) * (2 * HID);
  if (tid < 16) e_acc[tid] = 0.0f;
#if USE_TDM
  if (tid < 32) {
    tdm_load_2d_f32((unsigned)(uintptr_t)(void*)encF, src,
                    2 * HID, 16, 2 * HID, 2 * HID, 16);
    __builtin_amdgcn_s_wait_tensorcnt(0);
  }
#else
  for (int i = tid; i < 16 * 2 * HID; i += blockDim.x) encF[i] = src[i];
#endif
  __syncthreads();

  const int w = tid >> 5, lane = tid & 31;
  const int ml = lane & 15, half = lane >> 4;
  const float* arow = encF + (size_t)ml * (2 * HID);
  for (int nt = w; nt < 16; nt += 4) {
    const int gn = nt * 16 + ml;
    const float* bcol = attn_W + (size_t)gn * (3 * HID);  // W_enc = attn_W[:,0:512]
    v8f acc = {};
    acc = bf16_tile_mma(2 * HID, acc,
                        [&](int k) { return (__bf16)arow[k]; },
                        [&](int k) { return (__bf16)bcol[k]; });
    const float hp = hproj[(size_t)b * HID + gn];
    const float vn = vvec[gn];
#pragma unroll
    for (int r = 0; r < 8; ++r) {
      const float t = tanhf(acc[r] + hp);
      atomicAdd(&e_acc[r + 8 * half], t * vn);   // ds_add_f32
    }
  }
  __syncthreads();
  if (tid < 16) e_out[(size_t)b * SLEN + st * 16 + tid] = e_acc[tid];
}

// ---------------------------------------------------------------------------
// K3: softmax over s (400) + context = attn @ enc  (per batch). enc rows are
// L2-resident after K2; prefetch ahead of the matvec stream.
// ---------------------------------------------------------------------------
__global__ void k_softmax_ctx(const float* __restrict__ enc,
                              const float* __restrict__ e_in,
                              __bf16* __restrict__ cat_bf) {
  __shared__ float attn[512];
  __shared__ float red[256];
  const int b = blockIdx.x, tid = threadIdx.x;   // 256 threads
  const float v0 = e_in[(size_t)b * SLEN + tid];
  const float v1 = (tid + 256 < SLEN) ? e_in[(size_t)b * SLEN + 256 + tid] : -1e30f;
  red[tid] = fmaxf(v0, v1);
  __syncthreads();
  for (int s = 128; s > 0; s >>= 1) {
    if (tid < s) red[tid] = fmaxf(red[tid], red[tid + s]);
    __syncthreads();
  }
  const float mx = red[0];
  __syncthreads();
  const float ex0 = __expf(v0 - mx);
  const float ex1 = (tid + 256 < SLEN) ? __expf(v1 - mx) : 0.0f;
  red[tid] = ex0 + ex1;
  __syncthreads();
  for (int s = 128; s > 0; s >>= 1) {
    if (tid < s) red[tid] += red[tid + s];
    __syncthreads();
  }
  const float inv = 1.0f / red[0];
  attn[tid] = ex0 * inv;
  attn[tid + 256] = ex1 * inv;
  __syncthreads();

  float acc0 = 0.0f, acc1 = 0.0f;
  const float* eb = enc + (size_t)b * SLEN * (2 * HID);
  for (int s = 0; s < SLEN; ++s) {
    const float a = attn[s];
    if (((s & 7) == 0) && (s + 8 < SLEN))
      __builtin_prefetch(eb + (size_t)(s + 8) * (2 * HID) + tid, 0, 0);
    acc0 += a * eb[(size_t)s * (2 * HID) + tid];
    acc1 += a * eb[(size_t)s * (2 * HID) + 256 + tid];
  }
  cat_bf[(size_t)b * (3 * HID) + tid]       = (__bf16)acc0;  // context[:,0:256]
  cat_bf[(size_t)b * (3 * HID) + 256 + tid] = (__bf16)acc1;  // context[:,256:512]
}

// ---------------------------------------------------------------------------
// K4: cat_project = cat @ proj_W^T + proj_b  (64x256, K=768) -> bf16.
// ---------------------------------------------------------------------------
__global__ void k_proj(const __bf16* __restrict__ cat_bf,
                       const float* __restrict__ proj_W,
                       const float* __restrict__ proj_b,
                       __bf16* __restrict__ catproj_bf) {
  const int w    = threadIdx.x >> 5;
  const int lane = threadIdx.x & 31;
  const int ml   = lane & 15;
  const int half = lane >> 4;
  for (int it = 0; it < 4; ++it) {
    const int t  = w * 4 + it;
    const int mt = t & 3, nt = t >> 2;
    const int gm = mt * 16 + ml, gn = nt * 16 + ml;
    const __bf16* arow = cat_bf + (size_t)gm * (3 * HID);
    const float*  bcol = proj_W + (size_t)gn * (3 * HID);
    v8f acc = {};
    acc = bf16_tile_mma(3 * HID, acc,
                        [&](int k) { return arow[k]; },
                        [&](int k) { return (__bf16)bcol[k]; });
    const float bias = proj_b[gn];
#pragma unroll
    for (int r = 0; r < 8; ++r)
      catproj_bf[(size_t)(mt * 16 + r + 8 * half) * HID + gn] =
          (__bf16)(acc[r] + bias);
  }
}

// ---------------------------------------------------------------------------
// K5: word_distribution = cat_project @ out_W^T + out_b   (64x50000, K=256).
// Two 16-col N-tiles per block: a 32x256 f32 out_W tile (32 KB) is DMA'd into
// LDS by the TDM, then 8 waves = 2 N-tiles x 4 M-tiles of WMMA. The 51 MB
// out_W stream is the bandwidth roofline here (~2.2 us at 23.3 TB/s).
// ---------------------------------------------------------------------------
__global__ void k_out(const __bf16* __restrict__ catproj_bf,
                      const float* __restrict__ out_W,
                      const float* __restrict__ out_b,
                      float* __restrict__ wd) {
  __shared__ float bF[32 * HID];    // 32 KB, [n_local][k] f32
  const int nt0 = blockIdx.x * 2;   // 3125 N-tiles total (50000/16)
  const int tid = threadIdx.x;      // 256 threads / 8 waves
  const int nrows = (nt0 + 2 <= 3125) ? 32 : 16;
  const float* src = out_W + (size_t)nt0 * 16 * HID;
#if USE_TDM
  if (tid < 32) {
    tdm_load_2d_f32((unsigned)(uintptr_t)(void*)bF, src,
                    HID, (unsigned)nrows, HID, HID, (unsigned)nrows);
    __builtin_amdgcn_s_wait_tensorcnt(0);
  }
#else
  for (int i = tid; i < nrows * HID; i += blockDim.x) bF[i] = src[i];
#endif
  __syncthreads();

  const int w = tid >> 5, lane = tid & 31;
  const int mt = w & 3;             // 4 M-tiles
  const int nt = nt0 + (w >> 2);    // 2 N-tiles
  if (nt >= 3125) return;
  const int ml = lane & 15, half = lane >> 4;
  const int gn = nt * 16 + ml;
  const __bf16* arow = catproj_bf + (size_t)(mt * 16 + ml) * HID;
  const float*  bcol = bF + (size_t)((w >> 2) * 16 + ml) * HID;
  v8f acc = {};
  acc = bf16_tile_mma(HID, acc,
                      [&](int k) { return arow[k]; },
                      [&](int k) { return (__bf16)bcol[k]; });
  const float bias = out_b[gn];
#pragma unroll
  for (int r = 0; r < 8; ++r)
    wd[(size_t)(mt * 16 + r + 8 * half) * VOCAB + gn] = acc[r] + bias;
}

// ---------------------------------------------------------------------------
extern "C" void kernel_launch(void* const* d_in, const int* in_sizes, int n_in,
                              void* d_out, int out_size, void* d_ws, size_t ws_size,
                              hipStream_t stream) {
  (void)in_sizes; (void)n_in; (void)out_size; (void)ws_size;

  const int*   word   = (const int*)  d_in[0];
  const float* hidden = (const float*)d_in[1];
  const float* cell   = (const float*)d_in[2];
  const float* enc    = (const float*)d_in[3];
  /* d_in[4] = max_source_len (== 400, unused) */
  const float* embed  = (const float*)d_in[5];
  const float* W_ih   = (const float*)d_in[6];
  const float* W_hh   = (const float*)d_in[7];
  const float* b_ih   = (const float*)d_in[8];
  const float* b_hh   = (const float*)d_in[9];
  const float* attn_W = (const float*)d_in[10];
  const float* attn_b = (const float*)d_in[11];
  const float* vvec   = (const float*)d_in[12];
  const float* proj_W = (const float*)d_in[13];
  const float* proj_b = (const float*)d_in[14];
  const float* out_W  = (const float*)d_in[15];
  const float* out_b  = (const float*)d_in[16];

  // Outputs flat in return order: word_distribution, h_t, c_t.
  float* wd = (float*)d_out;
  float* ht = wd + (size_t)BATCH * VOCAB;
  float* ct = ht + (size_t)BATCH * HID;

  // Workspace layout (bytes):
  char* ws = (char*)d_ws;
  float*  gates      = (float*) (ws + 0);                       // 64*1024*4 = 262144
  float*  hproj      = (float*) (ws + 262144);                  // 64*256*4  =  65536
  float*  e_buf      = (float*) (ws + 262144 + 65536);          // 64*400*4  = 102400
  __bf16* cat_bf     = (__bf16*)(ws + 262144 + 65536 + 102400); // 64*768*2  =  98304
  __bf16* catproj_bf = (__bf16*)(ws + 262144 + 65536 + 102400 + 98304); // 32768

  k_gates      <<<8,              256, 0, stream>>>(word, hidden, embed, W_ih, W_hh, gates);
  k_lstm_pw    <<<64,             256, 0, stream>>>(gates, cell, b_ih, b_hh, ht, ct, cat_bf);
  k_hproj      <<<1,              512, 0, stream>>>(cat_bf, attn_W, attn_b, hproj);
  k_scores     <<<dim3(25, 64),   128, 0, stream>>>(enc, attn_W, hproj, vvec, e_buf);
  k_softmax_ctx<<<64,             256, 0, stream>>>(enc, e_buf, cat_bf);
  k_proj       <<<1,              512, 0, stream>>>(cat_bf, proj_W, proj_b, catproj_bf);
  k_out        <<<1563,           256, 0, stream>>>(catproj_bf, out_W, out_b, wd);
}
